// TKANCell_31799937859755
// MI455X (gfx1250) — compile-verified, matrix operand order checked by reading, not run on hardware
//
#include <hip/hip_runtime.h>
#include <cstdint>
#include <cstddef>

// ---------------------------------------------------------------------------
// TKAN cell on MI455X (gfx1250): all GEMMs on v_wmma_f32_16x16x32_f16 with
// double-buffered LDS staging via global_load_async_to_lds_b128 (ASYNCcnt),
// software-pipelined 2-deep with compile-time buffer indices.
// ---------------------------------------------------------------------------

typedef __attribute__((ext_vector_type(16))) _Float16 v16h;
typedef __attribute__((ext_vector_type(8)))  float    v8f;

union Frag { v16h h; uint32_t u[8]; };

#define TKB 4096   // batch
#define TKI 256    // input features
#define TKH 512    // hidden
#define TKL 3      // sublayers

__device__ __forceinline__ float dev_sigmoid(float x) { return 1.0f / (1.0f + __expf(-x)); }

// ---------------------------------------------------------------------------
// WMMA GEMM: C(M,N) = A(M,K) * W(N,K)^T [+ bias], f16 in / f32 accum.
// Block = 128(M) x 64(N), 256 threads = 8 waves, each wave 32x32 (2x2 WMMA tiles).
// Double-buffered LDS; async global->LDS staging; K/32 must be even.
// Requires: M%128==0, N%64==0, K%64==0, A/W 16B aligned.
// ---------------------------------------------------------------------------
#define BM 128
#define BN 64
#define BKK 32
#define LDS_STRIDE 40   // 32 f16 payload + 8 pad; 80B row stride keeps b128 aligned

__global__ __launch_bounds__(256) void wmma_gemm_f16(
    const _Float16* __restrict__ A, const _Float16* __restrict__ W,
    float* __restrict__ C, _Float16* __restrict__ C16, int ld16,
    const float* __restrict__ bias, int M, int N, int K)
{
    __shared__ alignas(16) _Float16 sA[2][BM * LDS_STRIDE];
    __shared__ alignas(16) _Float16 sB[2][BN * LDS_STRIDE];

    const int t    = threadIdx.x;
    const int wave = t >> 5;
    const int lane = t & 31;
    const int half = lane >> 4;   // which 16-lane group
    const int r16  = lane & 15;
    const int wm   = wave >> 1;   // 0..3 -> 32-row slab
    const int wn   = wave & 1;    // 0..1 -> 32-col slab
    const int m0   = blockIdx.y * BM;
    const int n0   = blockIdx.x * BN;

    // Per-thread staging coordinates: A tile = 512 b128-chunks (2/thread), B = 256 (1/thread)
    const int rA0 = t >> 2,         cA0 = t & 3;
    const int rA1 = (t + 256) >> 2, cA1 = t & 3;
    const int rB  = t >> 2,         cB  = t & 3;
    const _Float16* pA0 = A + (size_t)(m0 + rA0) * K + cA0 * 8;
    const _Float16* pA1 = A + (size_t)(m0 + rA1) * K + cA1 * 8;
    const _Float16* pB  = W + (size_t)(n0 + rB)  * K + cB  * 8;
    // LDS destinations (32-bit LDS addresses), hoisted per buffer
    uint32_t lA0[2], lA1[2], lB[2];
#pragma unroll
    for (int b = 0; b < 2; ++b) {
        lA0[b] = (uint32_t)(uintptr_t)&sA[b][rA0 * LDS_STRIDE + cA0 * 8];
        lA1[b] = (uint32_t)(uintptr_t)&sA[b][rA1 * LDS_STRIDE + cA1 * 8];
        lB[b]  = (uint32_t)(uintptr_t)&sB[b][rB  * LDS_STRIDE + cB  * 8];
    }

    v8f acc[2][2] = {};

    // Issue one stage (3 x async b128 per thread) into buffer `buf`, then bump pointers.
    auto issue_stage = [&](int buf) {
        asm volatile("global_load_async_to_lds_b128 %0, %1, off"
                     :: "v"(lA0[buf]), "v"(pA0) : "memory");
        asm volatile("global_load_async_to_lds_b128 %0, %1, off"
                     :: "v"(lA1[buf]), "v"(pA1) : "memory");
        asm volatile("global_load_async_to_lds_b128 %0, %1, off"
                     :: "v"(lB[buf]), "v"(pB) : "memory");
        pA0 += BKK; pA1 += BKK; pB += BKK;
    };

    // Consume buffer `buf`: fragment gathers + 4 WMMAs.
    auto compute_stage = [&](int buf) {
        const uint32_t* sA32 = reinterpret_cast<const uint32_t*>(sA[buf]);
        const uint32_t* sB32 = reinterpret_cast<const uint32_t*>(sB[buf]);
        // dword d holds K pair at k = (d&3)*2 + half*8 + (d>>2)*16
        // -> dwords 0..3 / 8..11 (+half*4) are contiguous: lowers to 2x ds_load_b128.
        Frag a[2], b[2];
#pragma unroll
        for (int i = 0; i < 2; ++i) {
            int rowA = (wm * 32 + i * 16 + r16) * (LDS_STRIDE / 2);
            int rowB = (wn * 32 + i * 16 + r16) * (LDS_STRIDE / 2);
#pragma unroll
            for (int d = 0; d < 8; ++d) {
                int kd = (d & 3) + half * 4 + ((d >> 2) << 3);
                a[i].u[d] = sA32[rowA + kd];
                b[i].u[d] = sB32[rowB + kd];
            }
        }
#pragma unroll
        for (int i = 0; i < 2; ++i)
#pragma unroll
            for (int j = 0; j < 2; ++j)
                acc[i][j] = __builtin_amdgcn_wmma_f32_16x16x32_f16(
                    false, a[i].h, false, b[j].h, (short)0, acc[i][j], false, false);
    };

    // ---- 2-deep software pipeline, unrolled x2 (nsteps is even for all call sites) ----
    const int nsteps = K / BKK;
    issue_stage(0);
    for (int s = 0; s < nsteps - 2; s += 2) {
        issue_stage(1);
        asm volatile("s_wait_asynccnt 0x3" ::: "memory");
        __syncthreads();
        compute_stage(0);
        __syncthreads();
        issue_stage(0);
        asm volatile("s_wait_asynccnt 0x3" ::: "memory");
        __syncthreads();
        compute_stage(1);
        __syncthreads();
    }
    issue_stage(1);
    asm volatile("s_wait_asynccnt 0x3" ::: "memory");
    __syncthreads();
    compute_stage(0);
    __syncthreads();
    asm volatile("s_wait_asynccnt 0x0" ::: "memory");
    __syncthreads();
    compute_stage(1);

    // Epilogue: C/D layout -> lane(half,r16): VGPR r = (M = half*8 + r, N = r16)
#pragma unroll
    for (int i = 0; i < 2; ++i) {
        int mb = m0 + wm * 32 + i * 16 + half * 8;
#pragma unroll
        for (int j = 0; j < 2; ++j) {
            int nb = n0 + wn * 32 + j * 16 + r16;
            float bv = bias ? bias[nb] : 0.0f;
#pragma unroll
            for (int r = 0; r < 8; ++r) {
                float v = acc[i][j][r] + bv;
                C[(size_t)(mb + r) * N + nb] = v;
                if (C16) C16[(size_t)(mb + r) * ld16 + nb] = (_Float16)v;
            }
        }
    }
}

// ---------------------------------------------------------------------------
// KAN activation expansion: per element -> [silu(x), Bspline_0..7(x)] as f16,
// written at out[b*ldout + colOff + i*9 + c] (matches pack_kan weight order).
// Uniform grid: knots t_j = (j-3)*0.4 - 1, j=0..11 (GRID=5, ORDER=3).
// ---------------------------------------------------------------------------
__global__ void expand_kan(const float* __restrict__ X, _Float16* __restrict__ out,
                           int ldout, int colOff, int IN, int total)
{
    int idx = blockIdx.x * blockDim.x + threadIdx.x;
    if (idx >= total) return;
    int b = idx / IN, i = idx - b * IN;
    float x = X[idx];
    float s = x * dev_sigmoid(x);

    const float h = 0.4f;
    float bs[11];
#pragma unroll
    for (int j = 0; j < 11; ++j) {
        float tj = (float)(j - 3) * h - 1.0f;
        bs[j] = (x >= tj && x < tj + h) ? 1.0f : 0.0f;
    }
#pragma unroll
    for (int ord = 1; ord <= 3; ++ord) {
        float inv = 1.0f / ((float)ord * h);
#pragma unroll
        for (int j = 0; j + ord < 11; ++j) {
            float tj   = (float)(j - 3) * h - 1.0f;
            float tend = tj + (float)(ord + 1) * h;
            bs[j] = (x - tj) * inv * bs[j] + (tend - x) * inv * bs[j + 1];
        }
    }
    _Float16* dst = out + (size_t)b * ldout + colOff + i * 9;
    dst[0] = (_Float16)s;
#pragma unroll
    for (int c = 0; c < 8; ++c) dst[1 + c] = (_Float16)bs[c];
}

// Pack KAN weights: W[o, colOff + i*9]     = base[o,i]
//                   W[o, colOff + i*9+1+c] = spline[o,i,c]   (f32 -> f16)
__global__ void pack_kan(const float* __restrict__ base, const float* __restrict__ spline,
                         _Float16* __restrict__ Wout, int ldout, int colOff,
                         int IN, int total)
{
    int idx = blockIdx.x * blockDim.x + threadIdx.x;
    if (idx >= total) return;
    int o = idx / IN, i = idx - o * IN;
    _Float16* dst = Wout + (size_t)o * ldout + colOff + i * 9;
    dst[0] = (_Float16)base[idx];
    const float* sp = spline + (size_t)idx * 8;
#pragma unroll
    for (int c = 0; c < 8; ++c) dst[1 + c] = (_Float16)sp[c];
}

// Plain f32->f16 copy into a (rowOff, colOff) window of a row-major f16 matrix
__global__ void pack_plain(const float* __restrict__ src, _Float16* __restrict__ dst,
                           int ldout, int rowOff, int colOff, int K, int total)
{
    int idx = blockIdx.x * blockDim.x + threadIdx.x;
    if (idx >= total) return;
    int o = idx / K, k = idx - o * K;
    dst[(size_t)(rowOff + o) * ldout + colOff + k] = (_Float16)src[idx];
}

// f32 copy (for bias concat)
__global__ void copy_f32(const float* __restrict__ src, float* __restrict__ dst, int n)
{
    int idx = blockIdx.x * blockDim.x + threadIdx.x;
    if (idx < n) dst[idx] = src[idx];
}

// Gate elementwise: c_t = sig(f)*c_prev + sig(i)*tanh(g); also tanh(c_t)
__global__ void gates_ew(const float* __restrict__ pre, const float* __restrict__ c_prev,
                         float* __restrict__ c_out, float* __restrict__ tanhc, int total)
{
    int idx = blockIdx.x * blockDim.x + threadIdx.x;
    if (idx >= total) return;
    int b = idx >> 9, c = idx & 511;
    const float* pr = pre + (size_t)b * 1536;
    float f  = dev_sigmoid(pr[c]);
    float ii = dev_sigmoid(pr[512 + c]);
    float gg = tanhf(pr[1024 + c]);
    float ct = f * c_prev[idx] + ii * gg;
    c_out[idx] = ct;
    tanhc[idx] = tanhf(ct);
}

__global__ void hout_ew(const float* __restrict__ ogpre, const float* __restrict__ tanhc,
                        float* __restrict__ h, int total)
{
    int idx = blockIdx.x * blockDim.x + threadIdx.x;
    if (idx < total) h[idx] = dev_sigmoid(ogpre[idx]) * tanhc[idx];
}

// Row LayerNorm over 512 cols, 256 threads per row (2 cols/thread)
__global__ __launch_bounds__(256) void layernorm_rows(
    const float* __restrict__ X, const float* __restrict__ g,
    const float* __restrict__ bta, float* __restrict__ out)
{
    __shared__ float red[256];
    const int row = blockIdx.x;
    const int tid = threadIdx.x;
    const float* xr = X + (size_t)row * TKH;
    float v0 = xr[tid], v1 = xr[tid + 256];
    red[tid] = v0 + v1;
    __syncthreads();
    for (int off = 128; off > 0; off >>= 1) {
        if (tid < off) red[tid] += red[tid + off];
        __syncthreads();
    }
    float mu = red[0] * (1.0f / TKH);
    __syncthreads();
    float d0 = v0 - mu, d1 = v1 - mu;
    red[tid] = d0 * d0 + d1 * d1;
    __syncthreads();
    for (int off = 128; off > 0; off >>= 1) {
        if (tid < off) red[tid] += red[tid + off];
        __syncthreads();
    }
    float inv = rsqrtf(red[0] * (1.0f / TKH) + 1e-5f);
    float* orow = out + (size_t)row * TKH;
    orow[tid]       = d0 * inv * g[tid]       + bta[tid];
    orow[tid + 256] = d1 * inv * g[tid + 256] + bta[tid + 256];
}

// ---------------------------------------------------------------------------
extern "C" void kernel_launch(void* const* d_in, const int* in_sizes, int n_in,
                              void* d_out, int out_size, void* d_ws, size_t ws_size,
                              hipStream_t stream)
{
    (void)in_sizes; (void)n_in; (void)out_size; (void)ws_size;
    const float* x_t      = (const float*)d_in[0];
    const float* h_prev   = (const float*)d_in[1];
    const float* c_prev   = (const float*)d_in[2];
    const float* z_prev   = (const float*)d_in[3];
    const float* wx_base  = (const float*)d_in[4];
    const float* wx_spl   = (const float*)d_in[5];
    const float* wh_base  = (const float*)d_in[6];
    const float* wh_spl   = (const float*)d_in[7];
    const float* hz_base  = (const float*)d_in[8];
    const float* hz_spl   = (const float*)d_in[9];
    const float* phi_base = (const float*)d_in[10];
    const float* phi_spl  = (const float*)d_in[11];
    const float* hh_w     = (const float*)d_in[12];
    const float* hh_b     = (const float*)d_in[13];
    const float* ln_g     = (const float*)d_in[14];
    const float* ln_b     = (const float*)d_in[15];
    const float* f_w      = (const float*)d_in[16];
    const float* f_b      = (const float*)d_in[17];
    const float* i_w      = (const float*)d_in[18];
    const float* i_b      = (const float*)d_in[19];
    const float* g_w      = (const float*)d_in[20];
    const float* g_b      = (const float*)d_in[21];
    const float* o_w      = (const float*)d_in[22];
    const float* o_b      = (const float*)d_in[23];

    float* out_h = (float*)d_out;
    float* out_c = out_h + (size_t)TKB * TKH;
    float* out_z = out_c + (size_t)TKB * TKH;

    // -------- workspace carve-out (256B aligned blocks) --------
    char* ws = (char*)d_ws;
    size_t off = 0;
    auto alloc = [&](size_t bytes) -> void* {
        off = (off + 255) & ~size_t(255);
        void* p = ws + off;
        off += bytes;
        return p;
    };
    const int KS  = 9 * TKI + 9 * TKH;   // 6912  (s-stage K)
    const int KP  = 9 * TKH;             // 4608  (phi-stage K)
    const int KZ  = TKH + 9 * TKH;       // 5120  (z-pre K)
    const int KG  = TKI + TKH;           // 768   (gates K)
    const int KO  = TKL * TKH;           // 1536  (output-gate K)

    _Float16* Ws    = (_Float16*)alloc((size_t)TKL * TKH * KS * 2);
    _Float16* Wphi  = (_Float16*)alloc((size_t)TKL * TKH * KP * 2);
    _Float16* W2    = (_Float16*)alloc((size_t)TKL * TKH * KZ * 2);
    _Float16* Wg    = (_Float16*)alloc((size_t)3 * TKH * KG * 2);
    _Float16* Wo    = (_Float16*)alloc((size_t)TKH * KO * 2);
    float*    gbias = (float*)   alloc((size_t)3 * TKH * 4);
    _Float16* cat16 = (_Float16*)alloc((size_t)TKB * KG * 2);
    _Float16* Axz   = (_Float16*)alloc((size_t)TKB * KS * 2);
    _Float16* AsE   = (_Float16*)alloc((size_t)TKB * KP * 2);
    _Float16* A2    = (_Float16*)alloc((size_t)TKB * KZ * 2);
    _Float16* o16   = (_Float16*)alloc((size_t)TKB * KO * 2);
    float*    gpre  = (float*)   alloc((size_t)TKB * 3 * TKH * 4);
    float*    sF    = (float*)   alloc((size_t)TKB * TKH * 4);
    float*    oF    = (float*)   alloc((size_t)TKB * TKH * 4);
    float*    tmpF  = (float*)   alloc((size_t)TKB * TKH * 4);
    float*    tanhc = (float*)   alloc((size_t)TKB * TKH * 4);
    float*    ogpre = (float*)   alloc((size_t)TKB * TKH * 4);

    auto g1 = [](int n) { return dim3((n + 255) / 256); };
    const dim3 blk(256);

    // -------- weight packing --------
    for (int l = 0; l < TKL; ++l) {
        int nXI = TKH * TKI, nHH = TKH * TKH;
        pack_kan<<<g1(nXI), blk, 0, stream>>>(wx_base + (size_t)l * nXI, wx_spl + (size_t)l * nXI * 8,
                                              Ws + (size_t)l * TKH * KS, KS, 0, TKI, nXI);
        pack_kan<<<g1(nHH), blk, 0, stream>>>(wh_base + (size_t)l * nHH, wh_spl + (size_t)l * nHH * 8,
                                              Ws + (size_t)l * TKH * KS, KS, 9 * TKI, TKH, nHH);
        pack_kan<<<g1(nHH), blk, 0, stream>>>(phi_base + (size_t)l * nHH, phi_spl + (size_t)l * nHH * 8,
                                              Wphi + (size_t)l * TKH * KP, KP, 0, TKH, nHH);
        pack_plain<<<g1(nHH), blk, 0, stream>>>(hh_w + (size_t)l * nHH,
                                                W2 + (size_t)l * TKH * KZ, KZ, 0, 0, TKH, nHH);
        pack_kan<<<g1(nHH), blk, 0, stream>>>(hz_base + (size_t)l * nHH, hz_spl + (size_t)l * nHH * 8,
                                              W2 + (size_t)l * TKH * KZ, KZ, TKH, TKH, nHH);
    }
    pack_plain<<<g1(TKH * KG), blk, 0, stream>>>(f_w, Wg, KG, 0,        0, KG, TKH * KG);
    pack_plain<<<g1(TKH * KG), blk, 0, stream>>>(i_w, Wg, KG, TKH,      0, KG, TKH * KG);
    pack_plain<<<g1(TKH * KG), blk, 0, stream>>>(g_w, Wg, KG, 2 * TKH,  0, KG, TKH * KG);
    pack_plain<<<g1(TKH * KO), blk, 0, stream>>>(o_w, Wo, KO, 0, 0, KO, TKH * KO);
    copy_f32<<<g1(TKH), blk, 0, stream>>>(f_b, gbias, TKH);
    copy_f32<<<g1(TKH), blk, 0, stream>>>(i_b, gbias + TKH, TKH);
    copy_f32<<<g1(TKH), blk, 0, stream>>>(g_b, gbias + 2 * TKH, TKH);

    // -------- activations shared across sublayers --------
    pack_plain<<<g1(TKB * TKI), blk, 0, stream>>>(x_t,    cat16, KG, 0, 0,   TKI, TKB * TKI);
    pack_plain<<<g1(TKB * TKH), blk, 0, stream>>>(h_prev, cat16, KG, 0, TKI, TKH, TKB * TKH);
    expand_kan<<<g1(TKB * TKI), blk, 0, stream>>>(x_t, Axz, KS, 0, TKI, TKB * TKI);

    // -------- gates GEMM + c_t --------
    wmma_gemm_f16<<<dim3((3 * TKH) / BN, TKB / BM), blk, 0, stream>>>(
        cat16, Wg, gpre, nullptr, 0, gbias, TKB, 3 * TKH, KG);
    gates_ew<<<g1(TKB * TKH), blk, 0, stream>>>(gpre, c_prev, out_c, tanhc, TKB * TKH);

    // -------- sublayers --------
    for (int l = 0; l < TKL; ++l) {
        const float* zl = z_prev + (size_t)l * TKB * TKH;
        // s_t = [silu/bases(x), silu/bases(z)] @ [Wx|Wh]^T
        expand_kan<<<g1(TKB * TKH), blk, 0, stream>>>(zl, Axz, KS, 9 * TKI, TKH, TKB * TKH);
        wmma_gemm_f16<<<dim3(TKH / BN, TKB / BM), blk, 0, stream>>>(
            Axz, Ws + (size_t)l * TKH * KS, sF, nullptr, 0, nullptr, TKB, TKH, KS);
        // o_t = phi(s_t); also emit f16 copy into r_t concat buffer
        expand_kan<<<g1(TKB * TKH), blk, 0, stream>>>(sF, AsE, KP, 0, TKH, TKB * TKH);
        wmma_gemm_f16<<<dim3(TKH / BN, TKB / BM), blk, 0, stream>>>(
            AsE, Wphi + (size_t)l * TKH * KP, oF, o16 + (size_t)l * TKH, KO, nullptr, TKB, TKH, KP);
        // z_t = LN( [z, silu/bases(o)] @ [hh|hz]^T + hh_b )
        pack_plain<<<g1(TKB * TKH), blk, 0, stream>>>(zl, A2, KZ, 0, 0, TKH, TKB * TKH);
        expand_kan<<<g1(TKB * TKH), blk, 0, stream>>>(oF, A2, KZ, TKH, TKH, TKB * TKH);
        wmma_gemm_f16<<<dim3(TKH / BN, TKB / BM), blk, 0, stream>>>(
            A2, W2 + (size_t)l * TKH * KZ, tmpF, nullptr, 0, hh_b + (size_t)l * TKH, TKB, TKH, KZ);
        layernorm_rows<<<dim3(TKB), blk, 0, stream>>>(
            tmpF, ln_g + (size_t)l * TKH, ln_b + (size_t)l * TKH, out_z + (size_t)l * TKB * TKH);
    }

    // -------- output gate + h_t --------
    wmma_gemm_f16<<<dim3(TKH / BN, TKB / BM), blk, 0, stream>>>(
        o16, Wo, ogpre, nullptr, 0, o_b, TKB, TKH, KO);
    hout_ew<<<g1(TKB * TKH), blk, 0, stream>>>(ogpre, tanhc, out_h, TKB * TKH);
}
